// Model_31516470018466
// MI455X (gfx1250) — matrix operand order repaired and checked by reading, not compiled, not run on hardware
//
#include <hip/hip_runtime.h>
#include <hip/hip_bf16.h>

// ---------------- model constants ----------------
#define L_LAYERS 4
#define CDIM     228        // n_embed
#define HDIM     57         // head dim (CDIM / 4 lanes)
#define TSEQ     1024
#define NTOK     2048       // B*T
#define KP       256        // CDIM padded to K-multiple of 32
#define FC       912        // 4*C
#define KP2      928        // FC padded to multiple of 32
#define VOCAB    50257

typedef __attribute__((ext_vector_type(16))) _Float16 v16h;
typedef __attribute__((ext_vector_type(8)))  _Float16 v8h;
typedef __attribute__((ext_vector_type(8)))  float    v8f;

// =======================================================================
// WMMA GEMM:  D[M,N] (f32) = A[M,KPT] (f16, zero-padded K) * W[N,KPT]^T (f16)
// W stored row-major [N,K] == B matrix column-major in K  -> direct frag load.
// One wave computes a (16*MR) x 64 tile: MR A-fragments are reused against
// 4 B-fragments per K step (4*MR WMMAs), raising L2 arithmetic intensity.
// KPT is a compile-time K so the K loop fully unrolls and the scheduler can
// software-pipeline global_load_b128 clauses across K steps under the WMMAs.
// =======================================================================
template<int MR, int KPT>
__global__ void gemm_wmma_kernel(const _Float16* __restrict__ A,
                                 const _Float16* __restrict__ W,
                                 float* __restrict__ D,
                                 int M, int N, int ldd, int NT4)
{
    const int wave = blockIdx.x * (blockDim.x >> 5) + (threadIdx.x >> 5);
    const int lane = threadIdx.x & 31;
    const int total = (M / (16 * MR)) * NT4;
    if (wave >= total) return;
    const int mt = wave / NT4;
    const int nt = wave - mt * NT4;
    const int m0 = mt * (16 * MR);
    const int n0 = nt << 6;

    v8f acc[MR][4];
#pragma unroll
    for (int i = 0; i < MR; ++i)
#pragma unroll
        for (int j = 0; j < 4; ++j) acc[i][j] = (v8f){};

    // A fragment (16-bit A 16x32 layout):
    //   lane<16 : halves 0..7 -> K k0+0..7,  halves 8..15 -> K k0+16..23
    //   lane>=16: halves 0..7 -> K k0+8..15, halves 8..15 -> K k0+24..31
    const int kgrpA = (lane >> 4) << 3;     // 0 or 8 halves
    const int kgrpB = (lane >> 4) << 3;     // 0 or 8 halves
    const _Float16* Abase = A + (size_t)(m0 + (lane & 15)) * KPT + kgrpA;
    const _Float16* Wbase = W + (size_t)(n0 + (lane & 15)) * KPT + kgrpB;

#pragma unroll
    for (int k0 = 0; k0 < KPT; k0 += 32) {
        v16h a[MR];
#pragma unroll
        for (int i = 0; i < MR; ++i) {
            v8h alo = *(const v8h*)(Abase + (size_t)(16 * i) * KPT + k0);
            v8h ahi = *(const v8h*)(Abase + (size_t)(16 * i) * KPT + k0 + 16);
#pragma unroll
            for (int e = 0; e < 8; ++e) { a[i][e] = alo[e]; a[i][e + 8] = ahi[e]; }
        }
#pragma unroll
        for (int j = 0; j < 4; ++j) {
            v8h blo = *(const v8h*)(Wbase + (size_t)(16 * j) * KPT + k0);
            v8h bhi = *(const v8h*)(Wbase + (size_t)(16 * j) * KPT + k0 + 16);
            v16h b;
#pragma unroll
            for (int e = 0; e < 8; ++e) { b[e] = blo[e]; b[e + 8] = bhi[e]; }
#pragma unroll
            for (int i = 0; i < MR; ++i)
                acc[i][j] = __builtin_amdgcn_wmma_f32_16x16x32_f16(
                    false, a[i], false, b, (short)0, acc[i][j], false, false);
        }
    }

    // C/D layout: VGPR r -> row M = r (+8 for upper lane half), col = lane&15
    const int c0 = n0 + (lane & 15);
#pragma unroll
    for (int i = 0; i < MR; ++i) {
        const int rbase = m0 + 16 * i + ((lane >> 4) << 3);
#pragma unroll
        for (int r = 0; r < 8; ++r) {
            float* Drow = D + (size_t)(rbase + r) * ldd;
            if (c0      < N) Drow[c0]      = acc[i][0][r];
            if (c0 + 16 < N) Drow[c0 + 16] = acc[i][1][r];
            if (c0 + 32 < N) Drow[c0 + 32] = acc[i][2][r];
            if (c0 + 48 < N) Drow[c0 + 48] = acc[i][3][r];
        }
    }
}

// ---------------- f32 -> f16 padded conversion (weights & activations) ----
__global__ void conv_pad_kernel(const float* __restrict__ src, _Float16* __restrict__ dst,
                                int R, int K, int Rp, int Kp)
{
    int i = blockIdx.x * blockDim.x + threadIdx.x;
    int total = Rp * Kp;
    if (i >= total) return;
    int r = i / Kp, k = i - r * Kp;
    float v = (r < R && k < K) ? src[(size_t)r * K + k] : 0.f;
    dst[i] = (_Float16)v;
}

// transpose conversion for w_scan (out = in @ w  =>  W^T stored [d][c])
__global__ void conv_padT_kernel(const float* __restrict__ src, _Float16* __restrict__ dst,
                                 int Cc, int Rp, int Kp)
{
    int i = blockIdx.x * blockDim.x + threadIdx.x;
    int total = Rp * Kp;
    if (i >= total) return;
    int r = i / Kp, k = i - r * Kp;    // r = d (out row), k = c (in col)
    float v = (r < Cc && k < Cc) ? src[(size_t)k * Cc + r] : 0.f;
    dst[i] = (_Float16)v;
}

// ---------------- embedding gather ----------------
__global__ void embed_kernel(const int* __restrict__ idx, const float* __restrict__ wte,
                             float* __restrict__ x)
{
    int i = blockIdx.x * blockDim.x + threadIdx.x;
    if (i >= NTOK * CDIM) return;
    int tok = i / CDIM, c = i - tok * CDIM;
    x[i] = wte[(size_t)idx[tok] * CDIM + c];
}

// ---------------- LayerNorm (no bias), one wave per token ----------------
__global__ void ln_kernel(const float* __restrict__ x, const float* __restrict__ g,
                          float* __restrict__ h)
{
    int wv = threadIdx.x >> 5, lane = threadIdx.x & 31;
    int tok = blockIdx.x * (blockDim.x >> 5) + wv;
    if (tok >= NTOK) return;
    const float* xp = x + (size_t)tok * CDIM;
    float s = 0.f;
    for (int c = lane; c < CDIM; c += 32) s += xp[c];
#pragma unroll
    for (int o = 1; o < 32; o <<= 1) s += __shfl_xor(s, o, 32);
    float mean = s * (1.0f / CDIM);
    float vs = 0.f;
    for (int c = lane; c < CDIM; c += 32) { float d = xp[c] - mean; vs += d * d; }
#pragma unroll
    for (int o = 1; o < 32; o <<= 1) vs += __shfl_xor(vs, o, 32);
    float inv = rsqrtf(vs * (1.0f / CDIM) + 1e-5f);
    float* hp = h + (size_t)tok * CDIM;
    for (int c = lane; c < CDIM; c += 32) hp[c] = (xp[c] - mean) * inv * g[c];
}

// ---------------- activation f32 [NTOK,C] -> f16 [NTOK,KP] padded ----------
__global__ void act_conv_kernel(const float* __restrict__ src, _Float16* __restrict__ dst)
{
    int i = blockIdx.x * blockDim.x + threadIdx.x;
    if (i >= NTOK * KP) return;
    int tok = i >> 8, k = i & (KP - 1);
    float v = (k < CDIM) ? src[(size_t)tok * CDIM + k] : 0.f;
    dst[i] = (_Float16)v;
}

// (q*v) -> f16 padded
__global__ void mul_conv_kernel(const float* __restrict__ q, const float* __restrict__ v,
                                _Float16* __restrict__ dst)
{
    int i = blockIdx.x * blockDim.x + threadIdx.x;
    if (i >= NTOK * KP) return;
    int tok = i >> 8, k = i & (KP - 1);
    float val = 0.f;
    if (k < CDIM) { size_t o = (size_t)tok * CDIM + k; val = q[o] * v[o]; }
    dst[i] = (_Float16)val;
}

// scan step: s = left(off) + right -> f16 padded; left uses ident for t < off
__global__ void scan_sum_kernel(const float* __restrict__ xs, const float* __restrict__ ident,
                                _Float16* __restrict__ dst, int off)
{
    int i = blockIdx.x * blockDim.x + threadIdx.x;
    if (i >= NTOK * KP) return;
    int tok = i >> 8, k = i & (KP - 1);
    float s = 0.f;
    if (k < CDIM) {
        int t = tok & (TSEQ - 1);
        float right = xs[(size_t)tok * CDIM + k];
        float left  = (t >= off) ? xs[(size_t)(tok - off) * CDIM + k] : ident[k];
        s = left + right;
    }
    dst[i] = (_Float16)s;
}

// gelu(mid) -> f16 [NTOK,KP2] padded
__global__ void gelu_conv_kernel(const float* __restrict__ mid, _Float16* __restrict__ dst)
{
    int i = blockIdx.x * blockDim.x + threadIdx.x;
    if (i >= NTOK * KP2) return;
    int tok = i / KP2, k = i - tok * KP2;
    float v = 0.f;
    if (k < FC) {
        float x = mid[(size_t)tok * FC + k];
        v = 0.5f * x * (1.0f + tanhf(0.7978845608f * (x + 0.044715f * x * x * x)));
    }
    dst[i] = (_Float16)v;
}

__global__ void add_kernel(float* __restrict__ x, const float* __restrict__ y, int n)
{
    int i = blockIdx.x * blockDim.x + threadIdx.x;
    if (i < n) x[i] += y[i];
}

// ---------------- scan combine: one wave32 per token ----------------
// q=left, k=right, v=vmix; 4x4 lane attention, rms, /(1+l), out = left + z
__global__ void combine_kernel(const float* __restrict__ xs, const float* __restrict__ vmix,
                               const float* __restrict__ ident, float* __restrict__ xn, int off)
{
    __shared__ float sL[8][CDIM];
    __shared__ float sR[8][CDIM];
    __shared__ float sV[8][CDIM];
    __shared__ float sAtt[8][16];

    int wv = threadIdx.x >> 5, lane = threadIdx.x & 31;
    int tok = blockIdx.x * 8 + wv;
    if (tok >= NTOK) return;
    int t = tok & (TSEQ - 1);

    const float* rp = xs + (size_t)tok * CDIM;
    const float* lp = (t >= off) ? xs + (size_t)(tok - off) * CDIM : ident;
    const float* vp = vmix + (size_t)tok * CDIM;
    for (int c = lane; c < CDIM; c += 32) {
        sL[wv][c] = lp[c]; sR[wv][c] = rp[c]; sV[wv][c] = vp[c];
    }
    // single wave owns this LDS slice: compiler-inserted DS waits suffice

    // scores + softmax over m (groups of 4 lanes = one row l)
    float sc = 0.f;
    if (lane < 16) {
        int l = lane >> 2, m = lane & 3;
        const float* ql = &sL[wv][l * HDIM];
        const float* km = &sR[wv][m * HDIM];
        for (int d = 0; d < HDIM; ++d) sc += ql[d] * km[d];
        sc *= 0.13245323570650439f;   // 1/sqrt(57)
    }
    float mx = sc;
    mx = fmaxf(mx, __shfl_xor(mx, 1, 32));
    mx = fmaxf(mx, __shfl_xor(mx, 2, 32));
    float e = __expf(sc - mx);
    float sum = e;
    sum += __shfl_xor(sum, 1, 32);
    sum += __shfl_xor(sum, 2, 32);
    if (lane < 16) sAtt[wv][lane] = e / sum;

    // z = att @ v ; each lane covers d = lane, lane+32
    float z[4][2]; float ss[4];
#pragma unroll
    for (int l = 0; l < 4; ++l) {
        float a0 = sAtt[wv][4 * l + 0], a1 = sAtt[wv][4 * l + 1];
        float a2 = sAtt[wv][4 * l + 2], a3 = sAtt[wv][4 * l + 3];
        float acc = 0.f;
        z[l][0] = 0.f; z[l][1] = 0.f;
        int i = 0;
        for (int d = lane; d < HDIM; d += 32, ++i) {
            float zv = a0 * sV[wv][0 * HDIM + d] + a1 * sV[wv][1 * HDIM + d]
                     + a2 * sV[wv][2 * HDIM + d] + a3 * sV[wv][3 * HDIM + d];
            z[l][i] = zv;
            acc += zv * zv;
        }
        ss[l] = acc;
    }
#pragma unroll
    for (int l = 0; l < 4; ++l) {
        float s = ss[l];
#pragma unroll
        for (int o = 1; o < 32; o <<= 1) s += __shfl_xor(s, o, 32);
        ss[l] = s;
    }
    float* outp = xn + (size_t)tok * CDIM;
#pragma unroll
    for (int l = 0; l < 4; ++l) {
        float scale = rsqrtf(ss[l] * (1.0f / HDIM) + 1e-6f) / (float)(1 + l);
        int i = 0;
        for (int d = lane; d < HDIM; d += 32, ++i)
            outp[l * HDIM + d] = sL[wv][l * HDIM + d] + z[l][i] * scale;
    }
}

// =======================================================================
// host-side orchestration
// =======================================================================
template<int MR, int KPT>
static inline void launch_gemm(const _Float16* A, const _Float16* W, float* D,
                               int M, int N, int ldd, hipStream_t s)
{
    int NT4 = (N + 63) >> 6;
    int total = (M / (16 * MR)) * NT4;
    int blocks = (total + 7) >> 3;          // 8 waves / block (256 threads)
    gemm_wmma_kernel<MR, KPT><<<blocks, 256, 0, s>>>(A, W, D, M, N, ldd, NT4);
}

static inline size_t align_up(size_t x) { return (x + 255) & ~(size_t)255; }

extern "C" void kernel_launch(void* const* d_in, const int* in_sizes, int n_in,
                              void* d_out, int out_size, void* d_ws, size_t ws_size,
                              hipStream_t stream)
{
    (void)in_sizes; (void)n_in; (void)out_size; (void)ws_size;
    const int*   idx    = (const int*)d_in[0];
    const float* wte    = (const float*)d_in[1];
    const float* ln1g   = (const float*)d_in[2];
    const float* ln2g   = (const float*)d_in[3];
    const float* w_up   = (const float*)d_in[4];
    const float* w_v    = (const float*)d_in[5];
    const float* w_cp   = (const float*)d_in[6];
    const float* w_scan = (const float*)d_in[7];
    const float* ident  = (const float*)d_in[8];
    const float* w_fc   = (const float*)d_in[9];
    const float* w_pr   = (const float*)d_in[10];
    float* out = (float*)d_out;

    // ---- workspace carve-up ----
    char* p = (char*)d_ws;
    size_t off = 0;
    auto alloc = [&](size_t bytes) { void* r = p + off; off = align_up(off + bytes); return r; };

    const size_t NC = (size_t)NTOK * CDIM;
    float* xres = (float*)alloc(NC * 4);
    float* h    = (float*)alloc(NC * 4);
    float* vbuf = (float*)alloc(NC * 4);
    float* xs   = (float*)alloc(NC * 4);
    float* xn   = (float*)alloc(NC * 4);
    float* vmix = (float*)alloc(NC * 4);
    float* ybuf = (float*)alloc(NC * 4);
    float* mid  = (float*)alloc((size_t)NTOK * FC * 4);

    _Float16* a16   = (_Float16*)alloc((size_t)NTOK * KP * 2);
    _Float16* mid16 = (_Float16*)alloc((size_t)NTOK * KP2 * 2);

    const int RP_C  = 256;    // ceil(228/64)*64
    const int RP_FC = 960;    // ceil(912/64)*64
    const int RP_V  = 50304;  // ceil(50257/64)*64
    _Float16 *wUp16[L_LAYERS], *wV16[L_LAYERS], *wCp16[L_LAYERS], *wS16[L_LAYERS],
             *wFc16[L_LAYERS], *wPr16[L_LAYERS];
    for (int l = 0; l < L_LAYERS; ++l) {
        wUp16[l] = (_Float16*)alloc((size_t)RP_C * KP * 2);
        wV16[l]  = (_Float16*)alloc((size_t)RP_C * KP * 2);
        wCp16[l] = (_Float16*)alloc((size_t)RP_C * KP * 2);
        wS16[l]  = (_Float16*)alloc((size_t)RP_C * KP * 2);
        wFc16[l] = (_Float16*)alloc((size_t)RP_FC * KP * 2);
        wPr16[l] = (_Float16*)alloc((size_t)RP_C * KP2 * 2);
    }
    _Float16* wte16 = (_Float16*)alloc((size_t)RP_V * KP * 2);

    const int TPB = 256;
    auto grid1 = [](size_t n) { return (unsigned)((n + 255) / 256); };

    // ---- weight conversions (every call; re-zeroes padding) ----
    for (int l = 0; l < L_LAYERS; ++l) {
        const size_t CC = (size_t)CDIM * CDIM;
        conv_pad_kernel<<<grid1((size_t)RP_C * KP), TPB, 0, stream>>>(w_up + l * CC, wUp16[l], CDIM, CDIM, RP_C, KP);
        conv_pad_kernel<<<grid1((size_t)RP_C * KP), TPB, 0, stream>>>(w_v  + l * CC, wV16[l],  CDIM, CDIM, RP_C, KP);
        conv_pad_kernel<<<grid1((size_t)RP_C * KP), TPB, 0, stream>>>(w_cp + l * CC, wCp16[l], CDIM, CDIM, RP_C, KP);
        conv_padT_kernel<<<grid1((size_t)RP_C * KP), TPB, 0, stream>>>(w_scan + l * CC, wS16[l], CDIM, RP_C, KP);
        conv_pad_kernel<<<grid1((size_t)RP_FC * KP), TPB, 0, stream>>>(w_fc + (size_t)l * FC * CDIM, wFc16[l], FC, CDIM, RP_FC, KP);
        conv_pad_kernel<<<grid1((size_t)RP_C * KP2), TPB, 0, stream>>>(w_pr + (size_t)l * CDIM * FC, wPr16[l], CDIM, FC, RP_C, KP2);
    }
    conv_pad_kernel<<<grid1((size_t)RP_V * KP), TPB, 0, stream>>>(wte, wte16, VOCAB, CDIM, RP_V, KP);

    // ---- embedding ----
    embed_kernel<<<grid1(NC), TPB, 0, stream>>>(idx, wte, xres);

    // ---- layers ----
    for (int l = 0; l < L_LAYERS; ++l) {
        ln_kernel<<<NTOK / 8, 256, 0, stream>>>(xres, ln1g + l * CDIM, h);
        act_conv_kernel<<<grid1((size_t)NTOK * KP), TPB, 0, stream>>>(h, a16);
        launch_gemm<2, KP>(a16, wUp16[l], xs,   NTOK, CDIM, CDIM, stream);   // qin -> xs
        launch_gemm<2, KP>(a16, wV16[l],  vbuf, NTOK, CDIM, CDIM, stream);   // v

        // Kogge-Stone scan over time (10 steps)
        float* cur = xs; float* oth = xn;
        for (int offt = 1; offt < TSEQ; offt <<= 1) {
            scan_sum_kernel<<<grid1((size_t)NTOK * KP), TPB, 0, stream>>>(cur, ident + l * CDIM, a16, offt);
            launch_gemm<2, KP>(a16, wS16[l], vmix, NTOK, CDIM, CDIM, stream);
            combine_kernel<<<NTOK / 8, 256, 0, stream>>>(cur, vmix, ident + l * CDIM, oth, offt);
            float* tmp = cur; cur = oth; oth = tmp;
        }

        mul_conv_kernel<<<grid1((size_t)NTOK * KP), TPB, 0, stream>>>(cur, vbuf, a16);
        launch_gemm<2, KP>(a16, wCp16[l], ybuf, NTOK, CDIM, CDIM, stream);
        add_kernel<<<grid1(NC), TPB, 0, stream>>>(xres, ybuf, (int)NC);

        ln_kernel<<<NTOK / 8, 256, 0, stream>>>(xres, ln2g + l * CDIM, h);
        act_conv_kernel<<<grid1((size_t)NTOK * KP), TPB, 0, stream>>>(h, a16);
        launch_gemm<2, KP>(a16, wFc16[l], mid, NTOK, FC, FC, stream);
        gelu_conv_kernel<<<grid1((size_t)NTOK * KP2), TPB, 0, stream>>>(mid, mid16);
        launch_gemm<2, KP2>(mid16, wPr16[l], ybuf, NTOK, CDIM, CDIM, stream);
        add_kernel<<<grid1(NC), TPB, 0, stream>>>(xres, ybuf, (int)NC);
    }

    // ---- LM head (dominant GEMM): [2048,228] @ wte^T -> [2048,50257] ----
    // 64x64 tile per wave (MR=4): 32 FLOP per L2 byte of fragment traffic;
    // fully-unrolled K (8 steps) lets loads pipeline under the WMMAs.
    act_conv_kernel<<<grid1((size_t)NTOK * KP), TPB, 0, stream>>>(xres, a16);
    launch_gemm<4, KP>(a16, wte16, out, NTOK, VOCAB, VOCAB, stream);
}
